// MambaBlock_20031727468551
// MI455X (gfx1250) — matrix operand order repaired and checked by reading, not compiled, not run on hardware
//
#include <hip/hip_runtime.h>
#include <hip/hip_bf16.h>
#include <math.h>

#define BSZ     4
#define LSEQ    2048
#define DMODEL  1024
#define DINNER  2048
#define NHEADS  32
#define HEADDIM 64
#define DSTATE  128
#define CONVDIM 2304            // DINNER + 2*DSTATE
#define DINPROJ 4384            // 2*DINNER + 2*DSTATE + NHEADS
#define NTOK    (BSZ * LSEQ)    // 8192

typedef __attribute__((ext_vector_type(16))) __bf16 v16bf;
typedef __attribute__((ext_vector_type(8)))  float  v8f;

union Frag16 { v16bf v; uint4 q[2]; unsigned short u[16]; };

// ---------------------------------------------------------------------------
// Direct hardware packed convert: D[15:0]=bf16(lo), D[31:16]=bf16(hi).
__device__ __forceinline__ unsigned int cvt_pk_bf16(float lo, float hi) {
    unsigned int d;
    asm("v_cvt_pk_bf16_f32 %0, %1, %2" : "=v"(d) : "v"(lo), "v"(hi));
    return d;
}

// Async global -> LDS copies (CDNA5, tracked on ASYNCcnt).
__device__ __forceinline__ void async_ld_b32(unsigned ldsaddr, const float* g) {
    asm volatile("global_load_async_to_lds_b32 %0, %1, off"
                 :: "v"(ldsaddr), "v"(g) : "memory");
}
__device__ __forceinline__ void async_ld_b128(unsigned ldsaddr, const unsigned short* g) {
    asm volatile("global_load_async_to_lds_b128 %0, %1, off"
                 :: "v"(ldsaddr), "v"(g) : "memory");
}
__device__ __forceinline__ void wait_async4() { asm volatile("s_wait_asynccnt 0x4" ::: "memory"); }
__device__ __forceinline__ void wait_async0() { asm volatile("s_wait_asynccnt 0x0" ::: "memory"); }

// ---------------------------------------------------------------------------
// Elementwise f32 -> bf16 (8 elements/thread, v_cvt_pk_bf16_f32 + b128 store).
__global__ void cvt_bf16_kernel(const float* __restrict__ src,
                                unsigned short* __restrict__ dst) {
    const size_t i = ((size_t)blockIdx.x * blockDim.x + threadIdx.x) * 8;
    const float4 a = *(const float4*)(src + i);
    const float4 b = *(const float4*)(src + i + 4);
    uint4 o;
    o.x = cvt_pk_bf16(a.x, a.y);
    o.y = cvt_pk_bf16(a.z, a.w);
    o.z = cvt_pk_bf16(b.x, b.y);
    o.w = cvt_pk_bf16(b.z, b.w);
    *(uint4*)(dst + i) = o;
}

// ---------------------------------------------------------------------------
// C[M,N] = A[M,K](bf16) * W[N,K](bf16)^T (+ residual[M,N] f32), f32 accum.
// WG tile 128x128, 8 wave32s, each wave 32(M)x64(N) = 2x4 v8f accums.
// Panels double-buffered in LDS, filled by global_load_async_to_lds_b128;
// panel i+1 DMA overlaps WMMA on panel i.  Ragged N handled by clamping the
// W row (uniform ASYNCcnt per wave) + guarded stores.
__launch_bounds__(256)
__global__ void wmma_gemm_bf16(const unsigned short* __restrict__ A,
                               const unsigned short* __restrict__ W,
                               const float* __restrict__ residual,
                               float* __restrict__ C,
                               int M, int N, int K) {
    __shared__ unsigned short As[2][128][32];   // 2 x 8 KB bf16 A panels
    __shared__ unsigned short Ws[2][128][32];   // 2 x 8 KB bf16 W panels

    const int tid  = threadIdx.x;
    const int m0   = blockIdx.y * 128;
    const int n0   = blockIdx.x * 128;
    const int wave = tid >> 5;
    const int lane = tid & 31;
    const int wm   = wave & 3;   // 4 waves along M -> 32 rows each
    const int wn   = wave >> 2;  // 2 waves along N -> 64 cols each

    v8f acc[2][4];
#pragma unroll
    for (int i = 0; i < 2; ++i)
#pragma unroll
        for (int j = 0; j < 4; ++j)
#pragma unroll
            for (int r = 0; r < 8; ++r) acc[i][j][r] = 0.0f;

    // staging map: thread -> (panel row lr, 16-half chunk lc); 2x b128 each
    const int lr = tid >> 1;
    const int lc = (tid & 1) * 16;
    const unsigned short* Ap = A + (size_t)(m0 + lr) * K + lc;
    int wr = n0 + lr; if (wr > N - 1) wr = N - 1;   // clamp ragged rows
    const unsigned short* Wp = W + (size_t)wr * K + lc;

    const unsigned ldsA0 = (unsigned)(uintptr_t)&As[0][lr][lc];
    const unsigned ldsA1 = (unsigned)(uintptr_t)&As[1][lr][lc];
    const unsigned ldsW0 = (unsigned)(uintptr_t)&Ws[0][lr][lc];
    const unsigned ldsW1 = (unsigned)(uintptr_t)&Ws[1][lr][lc];

    auto issue = [&](int k0, int s) {
        const unsigned la = s ? ldsA1 : ldsA0;
        const unsigned lw = s ? ldsW1 : ldsW0;
        async_ld_b128(la,      Ap + k0);
        async_ld_b128(la + 16, Ap + k0 + 8);
        async_ld_b128(lw,      Wp + k0);
        async_ld_b128(lw + 16, Wp + k0 + 8);
    };

    issue(0, 0);
    const int nk = K >> 5;
    const int am = lane & 15;
    const int kb = (lane >> 4) * 8;    // A frag K base (halfs)
    const int kbb = (lane >> 4) * 16;  // unused; B frag uses kb like A

    for (int i = 0; i < nk; ++i) {
        if (i + 1 < nk) { issue((i + 1) << 5, (i + 1) & 1); wait_async4(); }
        else            { wait_async0(); }
        __syncthreads();

        const unsigned short (*Asb)[32] = As[i & 1];
        const unsigned short (*Wsb)[32] = Ws[i & 1];

        // A fragment (ISA 16-bit A 16x32): lane m=l&15, K base=(l>=16)*8;
        // elems 0..7 = K kb..kb+7, elems 8..15 = K 16+kb..  -> 2x ds_load_b128
        Frag16 afr[2], bfr[4];
#pragma unroll
        for (int ii = 0; ii < 2; ++ii) {
            const int row = wm * 32 + ii * 16 + am;
            afr[ii].q[0] = *(const uint4*)&Asb[row][kb];
            afr[ii].q[1] = *(const uint4*)&Asb[row][16 + kb];
        }
        // B fragment = W^T tile: lane n=l&15, same K order as A fragment
#pragma unroll
        for (int j = 0; j < 4; ++j) {
            const int row = wn * 64 + j * 16 + am;
            bfr[j].q[0] = *(const uint4*)&Wsb[row][kb];
            bfr[j].q[1] = *(const uint4*)&Wsb[row][16 + kb];
        }
#pragma unroll
        for (int ii = 0; ii < 2; ++ii)
#pragma unroll
            for (int j = 0; j < 4; ++j)
                acc[ii][j] = __builtin_amdgcn_wmma_f32_16x16x32_bf16(
                    false, afr[ii].v, false, bfr[j].v,
                    (short)0, acc[ii][j], false, false);
        __syncthreads();
    }

    // C/D layout: VGPR r -> M=r (lanes 0-15) / r+8 (lanes 16-31); lane&15 -> N
    const int mo = (lane >> 4) * 8;
    const int nl = lane & 15;
#pragma unroll
    for (int i = 0; i < 2; ++i) {
#pragma unroll
        for (int j = 0; j < 4; ++j) {
            const int n = n0 + wn * 64 + j * 16 + nl;
            if (n >= N) continue;
#pragma unroll
            for (int r = 0; r < 8; ++r) {
                const int m = m0 + wm * 32 + i * 16 + mo + r;
                float v = acc[i][j][r];
                if (residual) v += residual[(size_t)m * N + n];
                C[(size_t)m * N + n] = v;
            }
        }
    }
}

// ---------------------------------------------------------------------------
// Depthwise causal conv(4) + SiLU over xBC channels; dt=softplus(.+bias),
// dA = exp(dt * -exp(A_log)).  One workgroup per token.
__global__ void conv_dt_kernel(const float* __restrict__ zxbcdt,
                               const float* __restrict__ conv_w,
                               const float* __restrict__ conv_b,
                               const float* __restrict__ dt_bias,
                               const float* __restrict__ A_log,
                               float* __restrict__ xconv,
                               float* __restrict__ dtb,
                               float* __restrict__ dab) {
    const int t = blockIdx.x;            // b*LSEQ + l
    const int l = t & (LSEQ - 1);
    for (int c = threadIdx.x; c < CONVDIM; c += blockDim.x) {
        float accv = conv_b[c];
#pragma unroll
        for (int w = 0; w < 4; ++w) {
            const int lw = l - 3 + w;
            if (lw >= 0)
                accv += zxbcdt[(size_t)(t + lw - l) * DINPROJ + DINNER + c] *
                        conv_w[c * 4 + w];
        }
        xconv[(size_t)t * CONVDIM + c] = accv / (1.0f + expf(-accv));
    }
    if (threadIdx.x < NHEADS) {
        const int h = threadIdx.x;
        const float raw = zxbcdt[(size_t)t * DINPROJ + DINNER + CONVDIM + h] + dt_bias[h];
        const float dtv = (raw > 20.0f) ? raw : log1pf(expf(raw));
        const float Ah  = -expf(A_log[h]);
        dtb[(size_t)t * NHEADS + h] = dtv;
        dab[(size_t)t * NHEADS + h] = expf(dtv * Ah);
    }
}

// ---------------------------------------------------------------------------
// Sequential SSM scan. One workgroup per (b,h); head state 64x128 f32 lives
// in VGPRs (32 floats/thread); B/C/x rows staged in LDS per step via the
// async global->LDS engine (wave-uniform branches: waves 0-3 B, 4-7 C, 0-1 x).
__launch_bounds__(256)
__global__ void scan_kernel(const float* __restrict__ xconv,
                            const float* __restrict__ dtb,
                            const float* __restrict__ dab,
                            const float* __restrict__ Dvec,
                            float* __restrict__ y) {
    const int b   = blockIdx.x >> 5;
    const int h   = blockIdx.x & 31;
    const int tid = threadIdx.x;
    const int p     = tid >> 2;        // head dim 0..63
    const int ng    = tid & 3;         // state quarter
    const int nbase = ng * 32;
    __shared__ float sB[DSTATE], sC[DSTATE], sX[HEADDIM], sScal[2];

    const unsigned ldsB = (unsigned)(uintptr_t)&sB[tid & (DSTATE - 1)];
    const unsigned ldsC = (unsigned)(uintptr_t)&sC[tid & (DSTATE - 1)];
    const unsigned ldsX = (unsigned)(uintptr_t)&sX[tid & (HEADDIM - 1)];

    float hst[32];
#pragma unroll
    for (int j = 0; j < 32; ++j) hst[j] = 0.0f;
    const float Dh = Dvec[h];

    for (int l = 0; l < LSEQ; ++l) {
        const int t = b * LSEQ + l;
        const size_t rowc = (size_t)t * CONVDIM;
        if (tid < DSTATE)  async_ld_b32(ldsB, xconv + rowc + DINNER + tid);
        else               async_ld_b32(ldsC, xconv + rowc + DINNER + DSTATE + (tid - DSTATE));
        if (tid < HEADDIM) async_ld_b32(ldsX, xconv + rowc + h * HEADDIM + tid);
        if (tid == 0) {
            sScal[0] = dtb[(size_t)t * NHEADS + h];
            sScal[1] = dab[(size_t)t * NHEADS + h];
        }
        wait_async0();
        __syncthreads();
        const float dtv = sScal[0], dAv = sScal[1];
        const float xv   = sX[p];
        const float coef = dtv * xv;
        float accv = 0.0f;
#pragma unroll
        for (int j = 0; j < 32; ++j) {
            const float hv = fmaf(hst[j], dAv, coef * sB[nbase + j]);
            hst[j] = hv;
            accv = fmaf(hv, sC[nbase + j], accv);
        }
        accv += __shfl_xor(accv, 1, 32);
        accv += __shfl_xor(accv, 2, 32);
        if (ng == 0)
            y[(size_t)t * DINNER + h * HEADDIM + p] = fmaf(Dh, xv, accv);
        __syncthreads();
    }
}

// ---------------------------------------------------------------------------
__device__ __forceinline__ float block_sum(float v, float* sbuf) {
#pragma unroll
    for (int off = 16; off > 0; off >>= 1) v += __shfl_xor(v, off, 32);
    const int lane = threadIdx.x & 31, w = threadIdx.x >> 5;
    if (lane == 0) sbuf[w] = v;
    __syncthreads();
    float r = (lane < 8) ? sbuf[lane] : 0.0f;
#pragma unroll
    for (int off = 16; off > 0; off >>= 1) r += __shfl_xor(r, off, 32);
    __syncthreads();
    return r;
}

// yb(bf16) = (y * silu(z)) * rsqrt(mean(.^2)+eps) * norm_w
// Contiguous 8 channels/thread; bf16 out feeds the out_proj WMMA GEMM.
__global__ void gate_rmsnorm_kernel(const float* __restrict__ zxbcdt,
                                    const float* __restrict__ norm_w,
                                    const float* __restrict__ y,
                                    unsigned short* __restrict__ yb) {
    const int t = blockIdx.x;
    const float* zrow = zxbcdt + (size_t)t * DINPROJ;
    const float* yrow = y + (size_t)t * DINNER;
    __shared__ float sbuf[8];
    const int c0 = threadIdx.x * 8;
    float vals[8], ss = 0.0f;
#pragma unroll
    for (int j = 0; j < 8; ++j) {
        const float z = zrow[c0 + j];
        const float v = yrow[c0 + j] * (z / (1.0f + expf(-z)));
        vals[j] = v;
        ss = fmaf(v, v, ss);
    }
    const float tot = block_sum(ss, sbuf);
    const float rms = rsqrtf(tot / (float)DINNER + 1e-5f);
    float o[8];
#pragma unroll
    for (int j = 0; j < 8; ++j) o[j] = vals[j] * rms * norm_w[c0 + j];
    uint4 pk;
    pk.x = cvt_pk_bf16(o[0], o[1]);
    pk.y = cvt_pk_bf16(o[2], o[3]);
    pk.z = cvt_pk_bf16(o[4], o[5]);
    pk.w = cvt_pk_bf16(o[6], o[7]);
    *(uint4*)(yb + (size_t)t * DINNER + c0) = pk;
}

// In-place LayerNorm over d_out rows of DMODEL.
__global__ void layernorm_kernel(const float* __restrict__ ln_w,
                                 const float* __restrict__ ln_b,
                                 float* __restrict__ out) {
    const int t = blockIdx.x;
    float* row = out + (size_t)t * DMODEL;
    __shared__ float sbuf[8];
    float vals[4], s = 0.0f, s2 = 0.0f;
#pragma unroll
    for (int j = 0; j < 4; ++j) {
        const int c = j * 256 + threadIdx.x;
        const float v = row[c];
        vals[j] = v;
        s += v;
        s2 = fmaf(v, v, s2);
    }
    const float mu  = block_sum(s, sbuf) / (float)DMODEL;
    const float var = block_sum(s2, sbuf) / (float)DMODEL - mu * mu;
    const float inv = rsqrtf(var + 1e-5f);
#pragma unroll
    for (int j = 0; j < 4; ++j) {
        const int c = j * 256 + threadIdx.x;
        row[c] = (vals[j] - mu) * inv * ln_w[c] + ln_b[c];
    }
}

// ---------------------------------------------------------------------------
extern "C" void kernel_launch(void* const* d_in, const int* in_sizes, int n_in,
                              void* d_out, int out_size, void* d_ws, size_t ws_size,
                              hipStream_t stream) {
    (void)in_sizes; (void)n_in; (void)out_size; (void)ws_size;
    const float* x          = (const float*)d_in[0];
    const float* in_proj_w  = (const float*)d_in[1];
    const float* conv_w     = (const float*)d_in[2];
    const float* conv_b     = (const float*)d_in[3];
    const float* dt_bias    = (const float*)d_in[4];
    const float* A_log      = (const float*)d_in[5];
    const float* Dv         = (const float*)d_in[6];
    const float* norm_w     = (const float*)d_in[7];
    const float* out_proj_w = (const float*)d_in[8];
    const float* ln_w       = (const float*)d_in[9];
    const float* ln_b       = (const float*)d_in[10];
    float* out = (float*)d_out;

    char* ws = (char*)d_ws;
    size_t off = 0;
    float* zxbcdt = (float*)(ws + off); off += (size_t)NTOK * DINPROJ * sizeof(float);
    float* xconv  = (float*)(ws + off); off += (size_t)NTOK * CONVDIM * sizeof(float);
    float* dtb    = (float*)(ws + off); off += (size_t)NTOK * NHEADS  * sizeof(float);
    float* dab    = (float*)(ws + off); off += (size_t)NTOK * NHEADS  * sizeof(float);
    float* ybuf   = (float*)(ws + off); off += (size_t)NTOK * DINNER  * sizeof(float);
    unsigned short* xb   = (unsigned short*)(ws + off); off += (size_t)NTOK * DMODEL * 2;
    unsigned short* ipwb = (unsigned short*)(ws + off); off += (size_t)DINPROJ * DMODEL * 2;
    unsigned short* opwb = (unsigned short*)(ws + off); off += (size_t)DMODEL * DINNER * 2;
    unsigned short* yb   = (unsigned short*)(ws + off); off += (size_t)NTOK * DINNER * 2;

    const dim3 blk(256);
    // 0) one-time f32 -> bf16 conversion of GEMM operands
    cvt_bf16_kernel<<<dim3((NTOK * DMODEL) / 2048), blk, 0, stream>>>(x, xb);
    cvt_bf16_kernel<<<dim3((DINPROJ * DMODEL) / 2048), blk, 0, stream>>>(in_proj_w, ipwb);
    cvt_bf16_kernel<<<dim3((DMODEL * DINNER) / 2048), blk, 0, stream>>>(out_proj_w, opwb);
    // 1) in_proj GEMM: zxbcdt[8192,4384] = x @ in_proj_w^T
    wmma_gemm_bf16<<<dim3((DINPROJ + 127) / 128, NTOK / 128), blk, 0, stream>>>(
        xb, ipwb, nullptr, zxbcdt, NTOK, DINPROJ, DMODEL);
    // 2) conv + SiLU + dt/dA
    conv_dt_kernel<<<dim3(NTOK), blk, 0, stream>>>(
        zxbcdt, conv_w, conv_b, dt_bias, A_log, xconv, dtb, dab);
    // 3) sequential SSM scan, one WG per (b,h)
    scan_kernel<<<dim3(BSZ * NHEADS), blk, 0, stream>>>(xconv, dtb, dab, Dv, ybuf);
    // 4) gate + RMSNorm -> bf16
    gate_rmsnorm_kernel<<<dim3(NTOK), blk, 0, stream>>>(zxbcdt, norm_w, ybuf, yb);
    // 5) out_proj GEMM + residual: out = yb @ out_proj_w^T + x
    wmma_gemm_bf16<<<dim3(DMODEL / 128, NTOK / 128), blk, 0, stream>>>(
        yb, opwb, x, out, NTOK, DMODEL, DINNER);
    // 6) LayerNorm (in place on d_out)
    layernorm_kernel<<<dim3(NTOK), blk, 0, stream>>>(ln_w, ln_b, out);
}
